// ResidualBlock_13443247636813
// MI455X (gfx1250) — matrix operand order repaired and checked by reading, not compiled, not run on hardware
//
#include <hip/hip_runtime.h>
#include <hip/hip_bf16.h>

#define EPSV 1e-5f

typedef __attribute__((ext_vector_type(16))) __bf16          v16bf;
typedef __attribute__((ext_vector_type(8)))  float           v8f;
typedef __attribute__((ext_vector_type(8)))  unsigned short  v8us;

// ---------------------------------------------------------------- helpers

static __device__ __forceinline__ unsigned short f2bfbits(float f) {
    unsigned u = __builtin_bit_cast(unsigned, f);
    unsigned r = u + 0x7FFFu + ((u >> 16) & 1u);
    return (unsigned short)(r >> 16);
}

static __device__ __forceinline__ v8f wmma_bf16(v16bf a, v16bf b, v8f c) {
    return __builtin_amdgcn_wmma_f32_16x16x32_bf16(false, a, false, b, (short)0, c,
                                                   false, false);
}

// ---- Fragment-swizzled LDS layouts -------------------------------------
// A 64x64 weight matrix lives in LDS as 8 fragments (kt=0..1, nt=0..3),
// each fragment = 32 lanes x 16 contiguous bf16 halves (1KB). A lane's
// whole v16bf fragment is one contiguous 32B block -> 2x ds_load_b128.
//
// B-fragment element i for lane l is  W[(kt*32 + (l>>4)*8 + i + (i<8?0:8))*64 + nt*16 + (l&15)].
// A 16x64 activation tile lives as 2 fragments (kt), same per-lane packing:
// A-fragment element i for lane l is  X[l&15][kt*32 + (l>>4)*8 + i + (i<8?0:8)].

static __device__ __forceinline__ v16bf load_frag_sw(const unsigned short* base,
                                                     int fragid, int lane) {
    const unsigned short* p = base + (fragid * 32 + lane) * 16;
    return *(const v16bf*)p;  // 32B contiguous, 32B aligned -> 2x b128
}

// Stage a row-major 64x64 f32 weight matrix into fragment-swizzled bf16 LDS.
// Dest-ordered: each thread writes 16B contiguous LDS stores.
static __device__ __forceinline__ void stage_weight_sw(unsigned short* dst,
                                                       const float* __restrict__ W,
                                                       int t, int nthreads) {
    for (int g = t; g < 512; g += nthreads) {
        int d0   = g * 8;               // half-index, 16B aligned
        int i0   = d0 & 15;             // 0 or 8
        int lane = (d0 >> 4) & 31;
        int fid  = d0 >> 9;
        int kt   = fid >> 2, nt = fid & 3;
        int n     = nt * 16 + (lane & 15);
        int kbase = kt * 32 + ((lane >> 4) << 3) + (i0 << 1);
        v8us tmp;
#pragma unroll
        for (int q = 0; q < 8; ++q) tmp[q] = f2bfbits(W[(kbase + q) * 64 + n]);
        *(v8us*)(dst + d0) = tmp;
    }
}

static __device__ __forceinline__ void zero_c(v8f* c) {
#pragma unroll
    for (int nt = 0; nt < 4; ++nt)
#pragma unroll
        for (int i = 0; i < 8; ++i) c[nt][i] = 0.0f;
}

// c[0..3] += (A tile, swizzled stage) @ (W, swizzled 64x64)
static __device__ __forceinline__ void gemm_stage(v8f* c, const unsigned short* stag,
                                                  const unsigned short* w, int lane) {
#pragma unroll
    for (int kt = 0; kt < 2; ++kt) {
        v16bf a = load_frag_sw(stag, kt, lane);
#pragma unroll
        for (int nt = 0; nt < 4; ++nt) {
            v16bf b = load_frag_sw(w, kt * 4 + nt, lane);
            c[nt]   = wmma_bf16(a, b, c[nt]);
        }
    }
}

// Store C-fragments (M=(lane>>4)*8+r, N=nt*16+(lane&15)) into an
// A-fragment-swizzled 16x64 stage tile (transpose -> scalar b16 stores).
static __device__ __forceinline__ void store_frag_lds(unsigned short* stag,
                                                      const v8f* f, int lane) {
    int n15 = lane & 15, half = lane >> 4;
    int laneadd = (n15 >> 3) << 4;          // 0 or 16
    int ibase   = n15 & 7;
#pragma unroll
    for (int nt = 0; nt < 4; ++nt) {
        int kt = nt >> 1;
        int i  = ibase + ((nt & 1) << 3);
#pragma unroll
        for (int r = 0; r < 8; ++r) {
            int M = half * 8 + r;
            stag[(kt * 32 + M + laneadd) * 16 + i] = f2bfbits(f[nt][r]);
        }
    }
}

// Per-row LayerNorm over 64 channels (16 lanes x 4 tiles) + opt residual + ReLU.
static __device__ __forceinline__ void ln_res_relu(v8f* f, const float* g,
                                                   const float* b, const v8f* res,
                                                   int n15, int useRes) {
#pragma unroll
    for (int r = 0; r < 8; ++r) {
        float s  = f[0][r] + f[1][r] + f[2][r] + f[3][r];
        float s2 = f[0][r] * f[0][r] + f[1][r] * f[1][r] +
                   f[2][r] * f[2][r] + f[3][r] * f[3][r];
#pragma unroll
        for (int m = 1; m < 16; m <<= 1) {
            s  += __shfl_xor(s, m, 32);
            s2 += __shfl_xor(s2, m, 32);
        }
        float mean = s * (1.0f / 64.0f);
        float var  = s2 * (1.0f / 64.0f) - mean * mean;
        float inv  = rsqrtf(var + EPSV);
#pragma unroll
        for (int nt = 0; nt < 4; ++nt) {
            int   Nc = nt * 16 + n15;
            float v  = (f[nt][r] - mean) * inv * g[Nc] + b[Nc];
            if (useRes) v += res[nt][r];
            f[nt][r] = fmaxf(v, 0.0f);
        }
    }
}

// ---------------------------------------------------------------- small kernels

__global__ void fill0_kernel(float* p, long n) {
    long i = (long)blockIdx.x * blockDim.x + threadIdx.x;
    if (i < n) p[i] = 0.0f;
}

__global__ void deg_kernel(const int* __restrict__ col, const float* __restrict__ ew,
                           float* deg, int E) {
    int e = blockIdx.x * blockDim.x + threadIdx.x;
    if (e < E) atomicAdd(&deg[col[e]], ew[e]);
}

__global__ void dinv_kernel(const float* __restrict__ deg, float* dinv, int n) {
    int i = blockIdx.x * blockDim.x + threadIdx.x;
    if (i < n) {
        float d = deg[i];
        dinv[i] = (d > 0.0f) ? rsqrtf(d) : 0.0f;
    }
}

__global__ void enorm_kernel(const int* __restrict__ row, const int* __restrict__ col,
                             const float* __restrict__ ew, const float* __restrict__ dinv,
                             float* nrm, int E) {
    int e = blockIdx.x * blockDim.x + threadIdx.x;
    if (e < E) nrm[e] = dinv[row[e]] * ew[e] * dinv[col[e]];
}

// Y[n,64] (+)= X[n,2] @ W[2,64]
__global__ void conv_c2_kernel(const float* __restrict__ X, const float* __restrict__ W,
                               float* Y, int n, int acc) {
    int t = blockIdx.x * blockDim.x + threadIdx.x;
    if (t >= n * 64) return;
    int ni = t >> 6, d = t & 63;
    float v = X[ni * 2 + 0] * W[d] + X[ni * 2 + 1] * W[64 + d];
    Y[t] = acc ? (Y[t] + v) : v;
}

// hdst[col[e], :] += norm[e] * hsrc[row[e], :]
__global__ void scatter_kernel(const float* __restrict__ hs, float* hd,
                               const int* __restrict__ row, const int* __restrict__ col,
                               const float* __restrict__ nrm, int E, int C, int chunks) {
    int t = blockIdx.x * blockDim.x + threadIdx.x;
    if (t >= E * chunks) return;
    int e  = t / chunks;
    int c0 = (t % chunks) * 16;
    int ce = (c0 + 16 < C) ? (c0 + 16) : C;
    float nv = nrm[e];
    long  rb = (long)row[e] * C, cb = (long)col[e] * C;
    for (int c = c0; c < ce; ++c) atomicAdd(&hd[cb + c], nv * hs[rb + c]);
}

// Y[N,64] (+)= X[N,64] @ W[64,64]  via WMMA bf16; 8 waves x 16 rows per block
__global__ __launch_bounds__(256) void gemm64_kernel(const float* __restrict__ X,
                                                     const float* __restrict__ Wsrc,
                                                     float* Y, int n, int acc) {
    __shared__ __align__(32) unsigned short sWb[4096];
    stage_weight_sw(sWb, Wsrc, threadIdx.x, 256);
    __syncthreads();

    int  wave = threadIdx.x >> 5, lane = threadIdx.x & 31;
    int  n15 = lane & 15, half = lane >> 4;
    long base = (long)blockIdx.x * 128 + wave * 16;

    long rrow  = base + n15;
    long rload = (rrow < n) ? rrow : (long)(n - 1);
    const float* xrow = X + rload * 64;

    v8f c[4];
#pragma unroll
    for (int nt = 0; nt < 4; ++nt)
#pragma unroll
        for (int r = 0; r < 8; ++r) {
            long M   = base + half * 8 + r;
            c[nt][r] = (acc && M < n) ? Y[M * 64 + nt * 16 + n15] : 0.0f;
        }

#pragma unroll
    for (int kt = 0; kt < 2; ++kt) {
        // A fragment: two contiguous 8-float K-runs per lane -> float4 loads
        int kb = half * 8;
        const float* p0 = xrow + kt * 32 + kb;
        const float* p1 = p0 + 16;
        float4 a0 = *(const float4*)(p0);
        float4 a1 = *(const float4*)(p0 + 4);
        float4 b0 = *(const float4*)(p1);
        float4 b1 = *(const float4*)(p1 + 4);
        float  run[16] = {a0.x, a0.y, a0.z, a0.w, a1.x, a1.y, a1.z, a1.w,
                          b0.x, b0.y, b0.z, b0.w, b1.x, b1.y, b1.z, b1.w};
        v16bf a;
#pragma unroll
        for (int i = 0; i < 16; ++i)
            a[i] = __builtin_bit_cast(__bf16, f2bfbits(run[i]));
#pragma unroll
        for (int nt = 0; nt < 4; ++nt) {
            v16bf b = load_frag_sw(sWb, kt * 4 + nt, lane);
            c[nt]   = wmma_bf16(a, b, c[nt]);
        }
    }

#pragma unroll
    for (int nt = 0; nt < 4; ++nt)
#pragma unroll
        for (int r = 0; r < 8; ++r) {
            long M = base + half * 8 + r;
            if (M < n) Y[M * 64 + nt * 16 + n15] = c[nt][r];
        }
}

// per-channel sum / sumsq over nodes
__global__ __launch_bounds__(256) void colstat_kernel(const float* __restrict__ X,
                                                      float* gs, float* gs2, int n) {
    __shared__ float ls[256], ls2[256];
    int   c = threadIdx.x & 63, g = threadIdx.x >> 6;
    float s = 0.0f, s2 = 0.0f;
    for (long ni = (long)blockIdx.x * 4 + g; ni < n; ni += (long)gridDim.x * 4) {
        float v = X[ni * 64 + c];
        s += v;
        s2 += v * v;
    }
    ls[threadIdx.x]  = s;
    ls2[threadIdx.x] = s2;
    __syncthreads();
    if (threadIdx.x < 64) {
        float a  = ls[c] + ls[64 + c] + ls[128 + c] + ls[192 + c];
        float a2 = ls2[c] + ls2[64 + c] + ls2[128 + c] + ls2[192 + c];
        atomicAdd(&gs[c], a);
        atomicAdd(&gs2[c], a2);
    }
}

__global__ void finstat_kernel(const float* gs, const float* gs2, float* mean,
                               float* inv, int n) {
    int c = threadIdx.x;
    if (c < 64) {
        float m = gs[c] / (float)n;
        float v = gs2[c] / (float)n - m * m;
        mean[c] = m;
        inv[c]  = rsqrtf(v + EPSV);
    }
}

// y = maybe_relu( maybe_res + maybe_inorm( x + maybe_bias ) )
__global__ void nodepost_kernel(const float* __restrict__ X, const float* __restrict__ bias,
                                const float* __restrict__ mean, const float* __restrict__ inv,
                                const float* __restrict__ res, float* Y, int n, int useBias,
                                int useNorm, int useRes, int useRelu) {
    long t = (long)blockIdx.x * blockDim.x + threadIdx.x;
    if (t >= (long)n * 64) return;
    int   d = (int)(t & 63);
    float v = X[t];
    if (useBias) v += bias[d];
    if (useNorm) v = (v - mean[d]) * inv[d];
    if (useRes) v += res[t];
    if (useRelu) v = fmaxf(v, 0.0f);
    Y[t] = v;
}

// ---------------------------------------------------------------- fused edge MLP

__global__ __launch_bounds__(64) void edge_mlp_kernel(
    const float* __restrict__ node, const int* __restrict__ row,
    const int* __restrict__ col, const float* __restrict__ ew,
    const float* __restrict__ We0, const float* __restrict__ be0,
    const float* __restrict__ ln0g, const float* __restrict__ ln0b,
    const float* __restrict__ Wa, const float* __restrict__ ba,
    const float* __restrict__ Wb, const float* __restrict__ bb,
    const float* __restrict__ lng, const float* __restrict__ lnb,
    const float* __restrict__ Wf, const float* __restrict__ bfv,
    float* __restrict__ out, float* __restrict__ esum, int E) {
    __shared__ __align__(32) unsigned short sW[6 * 4096];     // swizzled weights
    __shared__ __align__(32) unsigned short sStage[2][2][16 * 64];
    __shared__ float sVec[12 * 64];
    __shared__ float sWf[64];
    __shared__ float sEw[2][16];

    int t = threadIdx.x;
    stage_weight_sw(sW + 0 * 4096, We0,             t, 64);   // rows 0..63
    stage_weight_sw(sW + 1 * 4096, We0 + 64 * 64,   t, 64);   // rows 64..127
    stage_weight_sw(sW + 2 * 4096, Wa,              t, 64);   // Wa[0]
    stage_weight_sw(sW + 3 * 4096, Wb,              t, 64);   // Wb[0]
    stage_weight_sw(sW + 4 * 4096, Wa + 4096,       t, 64);   // Wa[1]
    stage_weight_sw(sW + 5 * 4096, Wb + 4096,       t, 64);   // Wb[1]
    if (t < 64) {
        sVec[0 * 64 + t]  = We0[128 * 64 + t];  // ew column of We0
        sVec[1 * 64 + t]  = be0[t];
        sVec[2 * 64 + t]  = ln0g[t];
        sVec[3 * 64 + t]  = ln0b[t];
        sVec[4 * 64 + t]  = ba[t];
        sVec[5 * 64 + t]  = bb[t];
        sVec[6 * 64 + t]  = ba[64 + t];
        sVec[7 * 64 + t]  = bb[64 + t];
        sVec[8 * 64 + t]  = lng[t];
        sVec[9 * 64 + t]  = lnb[t];
        sVec[10 * 64 + t] = lng[64 + t];
        sVec[11 * 64 + t] = lnb[64 + t];
        sWf[t]            = Wf[t];
    }

    int  wave = t >> 5, lane = t & 31;
    int  n15 = lane & 15, half = lane >> 4;
    long e0 = ((long)blockIdx.x * 2 + wave) * 16;

    // Gather endpoint features straight into A-fragment-swizzled stage tiles.
    // Lane handles edge (e0+n15), channels half*32..half*32+31; the four
    // 8-element K-runs are contiguous in fragment space -> 4x 16B ds stores.
    long eIdx = e0 + n15;
    long eCl  = (eIdx < E) ? eIdx : (long)(E - 1);
    int  rn = row[eCl], cn = col[eCl];
    {
        const float4* pr = (const float4*)(node + (long)rn * 64 + half * 32);
        const float4* pc = (const float4*)(node + (long)cn * 64 + half * 32);
        int base0 = (half * 32 + n15) * 16;        // lane n15,    i 0..7 / 8..15
        int base1 = (half * 32 + n15 + 16) * 16;   // lane n15+16, i 0..7 / 8..15
        unsigned short* dR = sStage[wave][0];
        unsigned short* dC = sStage[wave][1];
        float ar[32], ac[32];
#pragma unroll
        for (int q = 0; q < 8; ++q) {
            float4 a = pr[q];
            float4 b = pc[q];
            ar[q * 4 + 0] = a.x; ar[q * 4 + 1] = a.y;
            ar[q * 4 + 2] = a.z; ar[q * 4 + 3] = a.w;
            ac[q * 4 + 0] = b.x; ac[q * 4 + 1] = b.y;
            ac[q * 4 + 2] = b.z; ac[q * 4 + 3] = b.w;
        }
        v8us r0, r1, r2, r3, c0v, c1v, c2v, c3v;
#pragma unroll
        for (int j = 0; j < 8; ++j) {
            r0[j]  = f2bfbits(ar[j]);      r1[j]  = f2bfbits(ar[8 + j]);
            r2[j]  = f2bfbits(ar[16 + j]); r3[j]  = f2bfbits(ar[24 + j]);
            c0v[j] = f2bfbits(ac[j]);      c1v[j] = f2bfbits(ac[8 + j]);
            c2v[j] = f2bfbits(ac[16 + j]); c3v[j] = f2bfbits(ac[24 + j]);
        }
        *(v8us*)(dR + base0)     = r0;  *(v8us*)(dR + base1)     = r1;
        *(v8us*)(dR + base0 + 8) = r2;  *(v8us*)(dR + base1 + 8) = r3;
        *(v8us*)(dC + base0)     = c0v; *(v8us*)(dC + base1)     = c1v;
        *(v8us*)(dC + base0 + 8) = c2v; *(v8us*)(dC + base1 + 8) = c3v;
    }
    if (half == 0) sEw[wave][n15] = (eIdx < E) ? ew[eIdx] : 0.0f;
    __syncthreads();

    // Linear0: h = data[row]@Wr + data[col]@Wc + ew*w_ew + be0
    v8f h[4];
    zero_c(h);
    gemm_stage(h, sStage[wave][0], sW + 0 * 4096, lane);
    gemm_stage(h, sStage[wave][1], sW + 1 * 4096, lane);
#pragma unroll
    for (int nt = 0; nt < 4; ++nt)
#pragma unroll
        for (int r = 0; r < 8; ++r) {
            int M = half * 8 + r, Nc = nt * 16 + n15;
            h[nt][r] += sEw[wave][M] * sVec[Nc] + sVec[64 + Nc];
        }
    ln_res_relu(h, sVec + 2 * 64, sVec + 3 * 64, nullptr, n15, 0);
    __syncthreads();
    store_frag_lds(sStage[wave][0], h, lane);
    __syncthreads();

    // two residual blocks: h = relu( LN( relu(h@Wa+ba)@Wb+bb ) + h )
#pragma unroll
    for (int j = 0; j < 2; ++j) {
        v8f tt[4];
        zero_c(tt);
        gemm_stage(tt, sStage[wave][0], sW + (2 + 2 * j) * 4096, lane);
#pragma unroll
        for (int nt = 0; nt < 4; ++nt)
#pragma unroll
            for (int r = 0; r < 8; ++r) {
                int Nc    = nt * 16 + n15;
                tt[nt][r] = fmaxf(tt[nt][r] + sVec[(4 + 2 * j) * 64 + Nc], 0.0f);
            }
        __syncthreads();
        store_frag_lds(sStage[wave][1], tt, lane);
        __syncthreads();

        v8f u[4];
        zero_c(u);
        gemm_stage(u, sStage[wave][1], sW + (3 + 2 * j) * 4096, lane);
#pragma unroll
        for (int nt = 0; nt < 4; ++nt)
#pragma unroll
            for (int r = 0; r < 8; ++r)
                u[nt][r] += sVec[(5 + 2 * j) * 64 + nt * 16 + n15];
        ln_res_relu(u, sVec + (8 + 2 * j) * 64, sVec + (9 + 2 * j) * 64, h, n15, 1);
#pragma unroll
        for (int nt = 0; nt < 4; ++nt) h[nt] = u[nt];
        __syncthreads();
        store_frag_lds(sStage[wave][0], h, lane);
        __syncthreads();
    }

    // final Lin(64,1): o[M] = sum_N h[M,N]*Wf[N] + bf ; global stats via atomics
    float biasf = bfv[0];
    float lsum = 0.0f, lsum2 = 0.0f;
#pragma unroll
    for (int r = 0; r < 8; ++r) {
        float p = h[0][r] * sWf[0 * 16 + n15] + h[1][r] * sWf[1 * 16 + n15] +
                  h[2][r] * sWf[2 * 16 + n15] + h[3][r] * sWf[3 * 16 + n15];
#pragma unroll
        for (int m = 1; m < 16; m <<= 1) p += __shfl_xor(p, m, 32);
        float o = p + biasf;
        long  e = e0 + half * 8 + r;
        if (n15 == 0 && e < E) {
            out[e] = o;
            lsum += o;
            lsum2 += o * o;
        }
    }
    if (n15 == 0) {
        atomicAdd(&esum[0], lsum);
        atomicAdd(&esum[1], lsum2);
    }
}

__global__ void finalnorm_kernel(float* out, const float* __restrict__ esum, int E) {
    int e = blockIdx.x * blockDim.x + threadIdx.x;
    if (e >= E) return;
    float s1   = esum[0], s2 = esum[1];
    float mean = s1 / (float)E;
    float var  = (s2 - s1 * s1 / (float)E) / (float)(E - 1);
    float inv  = rsqrtf(var);
    out[e]     = fabsf((out[e] - mean) * inv);
}

// ---------------------------------------------------------------- launch

extern "C" void kernel_launch(void* const* d_in, const int* in_sizes, int n_in,
                              void* d_out, int out_size, void* d_ws, size_t ws_size,
                              hipStream_t stream) {
    const float* x      = (const float*)d_in[0];
    const int*   rowp   = (const int*)d_in[1];
    const float* ea     = (const float*)d_in[2];
    const float* tag0_w = (const float*)d_in[3];
    const float* tag0_b = (const float*)d_in[4];
    const float* tag_w  = (const float*)d_in[5];
    const float* tag_b  = (const float*)d_in[6];
    const float* We0    = (const float*)d_in[7];
    const float* be0    = (const float*)d_in[8];
    const float* ln0g   = (const float*)d_in[9];
    const float* ln0b   = (const float*)d_in[10];
    const float* Wa     = (const float*)d_in[11];
    const float* ba     = (const float*)d_in[12];
    const float* Wb     = (const float*)d_in[13];
    const float* bbv    = (const float*)d_in[14];
    const float* lng    = (const float*)d_in[15];
    const float* lnb    = (const float*)d_in[16];
    const float* Wf     = (const float*)d_in[17];
    const float* bfv    = (const float*)d_in[18];
    float*       outp   = (float*)d_out;

    const int  N    = in_sizes[0] / 2;
    const int  E    = in_sizes[2];
    const int* colp = rowp + E;

    float* ws   = (float*)d_ws;
    float* deg  = ws;
    float* dinv = deg + N;
    float* nrm  = dinv + N;
    float* data = nrm + E;
    size_t nd   = (size_t)N * 64;
    float* t1   = data + nd;
    float* hA   = t1 + nd;
    float* hB   = hA + nd;
    float* accb = hB + nd;
    float* gs   = accb + nd;
    float* gs2  = gs + 64;
    float* gmn  = gs2 + 64;
    float* ginv = gmn + 64;
    float* esum = ginv + 64;

#define NB(n) ((int)(((long)(n) + 255) / 256))
    const int gbGemm = (N + 127) / 128;

    // ---- GCN normalization
    fill0_kernel<<<NB(N), 256, 0, stream>>>(deg, N);
    deg_kernel<<<NB(E), 256, 0, stream>>>(colp, ea, deg, E);
    dinv_kernel<<<NB(N), 256, 0, stream>>>(deg, dinv, N);
    enorm_kernel<<<NB(E), 256, 0, stream>>>(rowp, colp, ea, dinv, nrm, E);

    // ---- TAGConv 0: IN_C=2 -> 64
    conv_c2_kernel<<<NB((long)N * 64), 256, 0, stream>>>(x, tag0_w, accb, N, 0);
    {
        const float* hcur = x;
        for (int k = 1; k <= 3; ++k) {
            float* ht = (k & 1) ? hA : hB;
            fill0_kernel<<<NB((long)N * 2), 256, 0, stream>>>(ht, (long)N * 2);
            scatter_kernel<<<NB(E), 256, 0, stream>>>(hcur, ht, rowp, colp, nrm, E, 2, 1);
            conv_c2_kernel<<<NB((long)N * 64), 256, 0, stream>>>(ht, tag0_w + k * 128,
                                                                 accb, N, 1);
            hcur = ht;
        }
    }
    fill0_kernel<<<1, 256, 0, stream>>>(gs, 128);
    colstat_kernel<<<256, 256, 0, stream>>>(accb, gs, gs2, N);
    finstat_kernel<<<1, 64, 0, stream>>>(gs, gs2, gmn, ginv, N);
    nodepost_kernel<<<NB(nd), 256, 0, stream>>>(accb, tag0_b, gmn, ginv, nullptr, data,
                                                N, 1, 1, 0, 1);

    // ---- D->D TAGConv helper
    auto conv64 = [&](const float* Xin, int l) {
        const float* Wl = tag_w + (size_t)l * 4 * 4096;
        gemm64_kernel<<<gbGemm, 256, 0, stream>>>(Xin, Wl, accb, N, 0);
        const float* hcur = Xin;
        for (int k = 1; k <= 3; ++k) {
            float* ht = (k & 1) ? hA : hB;
            fill0_kernel<<<NB(nd), 256, 0, stream>>>(ht, (long)nd);
            scatter_kernel<<<NB((long)E * 4), 256, 0, stream>>>(hcur, ht, rowp, colp,
                                                                nrm, E, 64, 4);
            gemm64_kernel<<<gbGemm, 256, 0, stream>>>(ht, Wl + k * 4096, accb, N, 1);
            hcur = ht;
        }
    };
    auto stats = [&](const float* X) {
        fill0_kernel<<<1, 256, 0, stream>>>(gs, 128);
        colstat_kernel<<<256, 256, 0, stream>>>(X, gs, gs2, N);
        finstat_kernel<<<1, 64, 0, stream>>>(gs, gs2, gmn, ginv, N);
    };

    // ---- residual blocks
    for (int i = 0; i < 2; ++i) {
        conv64(data, 2 * i);
        nodepost_kernel<<<NB(nd), 256, 0, stream>>>(accb, tag_b + (2 * i) * 64, gmn,
                                                    ginv, nullptr, t1, N, 1, 0, 0, 1);
        conv64(t1, 2 * i + 1);
        stats(accb);
        nodepost_kernel<<<NB(nd), 256, 0, stream>>>(accb, tag_b + (2 * i + 1) * 64, gmn,
                                                    ginv, data, data, N, 1, 1, 1, 1);
    }
    // ---- final TAGConv + residual (no norm)
    conv64(data, 4);
    nodepost_kernel<<<NB(nd), 256, 0, stream>>>(accb, tag_b + 4 * 64, gmn, ginv, data,
                                                data, N, 1, 0, 1, 1);

    // ---- fused edge MLP (WMMA) + stats
    fill0_kernel<<<1, 256, 0, stream>>>(esum, 2);
    edge_mlp_kernel<<<(E + 31) / 32, 64, 0, stream>>>(
        data, rowp, colp, ea, We0, be0, ln0g, ln0b, Wa, ba, Wb, bbv, lng, lnb, Wf, bfv,
        outp, esum, E);
    finalnorm_kernel<<<NB(E), 256, 0, stream>>>(outp, esum, E);
#undef NB
}
